// PointNetPlusPlusEncoder_25804163514716
// MI455X (gfx1250) — compile-verified, matrix operand order checked by reading, not run on hardware
//
#include <hip/hip_runtime.h>

// ---------------------------------------------------------------------------
// PointNet++ encoder for MI455X (gfx1250, wave32, WMMA).
// v_wmma_f32_16x16x32_f16 (f16 in / f32 accumulate). Weights (~1.8MB f16)
// are pre-swizzled into WMMA B-fragment order so each lane fetches its whole
// fragment as one contiguous 32B load (2x global_load_b128, L2-resident).
// Activations stay fused in LDS per workgroup; A-fragments are two aligned
// ds_load_b128 per tile.
// ---------------------------------------------------------------------------

typedef __attribute__((ext_vector_type(16))) _Float16 v16h;
typedef __attribute__((ext_vector_type(8)))  _Float16 v8h;
typedef __attribute__((ext_vector_type(8)))  float    v8f;

#define BB   16
#define PP   4096
#define OUTC 256
#define KSA  32
#define KFP  3

static const int M1 = PP / 4;   // 1024
static const int M2 = PP / 16;  // 256
static const int M3 = PP / 64;  // 64

// Layer tables (insertion order: sa1 sa2 sa3 fp3 fp2 fp1; W then b per layer)
static const int L_cin[15]  = {6, 64, 64, 131, 128, 128, 259, 256, 256, 768, 256, 384, 256, 259, 256};
static const int L_cout[15] = {64, 64, 128, 128, 128, 256, 256, 256, 512, 256, 256, 256, 256, 256, 256};
static const int L_cinp[15] = {32, 64, 64, 160, 128, 128, 288, 256, 256, 768, 256, 384, 256, 288, 256};

// ---------------------------------------------------------------------------
// WMMA fragment mapping (ISA 7.12.2, wave32).
// A (16x32 f16): lane L(0..15): M=L, holds K 0..7 (V0-3) and 16..23 (V4-7);
//                lane L+16:     M=L, holds K 8..15 and 24..31.
// B (32x16 f16): mirrored with N = lane&15.
// C/D (16x16 f32): lane: N = lane&15, element r: M = r + 8*(lane>=16).
// ---------------------------------------------------------------------------

// A-fragment from LDS (row-major, ld multiple of 32): two aligned 16B loads.
__device__ __forceinline__ v16h load_tile_a(const _Float16* buf, int row0, int k0, int ld) {
  const int lane  = threadIdx.x & 31;
  const int m     = lane & 15;
  const int khalf = (lane >> 4) & 1;
  const _Float16* rp = buf + (size_t)(row0 + m) * ld + k0 + khalf * 8;
  const v8h lo = *(const v8h*)(rp);       // K: kbase=0,  8 contiguous halfs
  const v8h hi = *(const v8h*)(rp + 16);  // K: kbase=16, 8 contiguous halfs
  return __builtin_shufflevector(lo, hi, 0, 1, 2, 3, 4, 5, 6, 7,
                                 8, 9, 10, 11, 12, 13, 14, 15);
}

// B-fragment from pre-swizzled weights: one contiguous 32B load per lane.
// Tile order: tile = (n0/16)*(CINP/32) + (k0/32); within a tile, lane-major
// blocks of 16 halfs already in fragment element order.
__device__ __forceinline__ v16h load_tile_b(const _Float16* W, int k0, int n0, int cinp) {
  const int lane = threadIdx.x & 31;
  const int tile = (n0 >> 4) * (cinp >> 5) + (k0 >> 5);
  return ((const v16h*)W)[(size_t)tile * 32 + lane];
}

// One MLP layer, executed per-wave on a 16-row stripe held in LDS.
// Output -> LDS as f16 (optional ReLU).
template <int CINP, int COUT>
__device__ __forceinline__ void mlp_layer_lds(const _Float16* inBuf, int row0,
                                              const _Float16* W, const float* bias,
                                              bool relu, _Float16* outBuf) {
  const int lane  = threadIdx.x & 31;
  const int col   = lane & 15;
  const int rbase = (lane >> 4) * 8;
#pragma unroll 1
  for (int n0 = 0; n0 < COUT; n0 += 16) {
    v8f acc = {};
#pragma unroll
    for (int k0 = 0; k0 < CINP; k0 += 32) {
      v16h a  = load_tile_a(inBuf, row0, k0, CINP);
      v16h bt = load_tile_b(W, k0, n0, CINP);
      acc = __builtin_amdgcn_wmma_f32_16x16x32_f16(false, a, false, bt,
                                                   (short)0, acc, false, false);
    }
    const float bv = bias[n0 + col];
#pragma unroll
    for (int r = 0; r < 8; ++r) {
      float v = acc[r] + bv;
      if (relu) v = fmaxf(v, 0.0f);
      outBuf[(size_t)(row0 + rbase + r) * COUT + (n0 + col)] = (_Float16)v;
    }
  }
}

// Final MLP layer: f32 straight to global (no ReLU). outG points at this
// block's first output row; row stride == COUT.
template <int CINP, int COUT>
__device__ __forceinline__ void mlp_layer_out(const _Float16* inBuf, int row0,
                                              const _Float16* W, const float* bias,
                                              float* outG) {
  const int lane  = threadIdx.x & 31;
  const int col   = lane & 15;
  const int rbase = (lane >> 4) * 8;
#pragma unroll 1
  for (int n0 = 0; n0 < COUT; n0 += 16) {
    v8f acc = {};
#pragma unroll
    for (int k0 = 0; k0 < CINP; k0 += 32) {
      v16h a  = load_tile_a(inBuf, row0, k0, CINP);
      v16h bt = load_tile_b(W, k0, n0, CINP);
      acc = __builtin_amdgcn_wmma_f32_16x16x32_f16(false, a, false, bt,
                                                   (short)0, acc, false, false);
    }
    const float bv = bias[n0 + col];
#pragma unroll
    for (int r = 0; r < 8; ++r) {
      outG[(size_t)(row0 + rbase + r) * COUT + (n0 + col)] = acc[r] + bv;
    }
  }
}

// ---------------------------------------------------------------------------
// Weight prep: f32 [Cin,Cout] row-major -> f16 fragment-major:
// tiles (nt, kt) in order tile = nt*(Cinp/32)+kt; per tile 32 lanes x 16
// halfs, element e of lane's v16h:
//   v=e/2, h=e%2, khalf=lane>>4, kbase=(v&4)?16:0
//   k = kt*32 + kbase + khalf*8 + (v&3)*2 + h ; n = nt*16 + (lane&15)
// Zero for k >= Cin (padding rows).
// ---------------------------------------------------------------------------
__global__ void wconv_kernel(const float* __restrict__ W, _Float16* __restrict__ Wp,
                             int cin, int cout, int cinp) {
  const int total = cinp * cout;  // == numTiles * 512
  int t = blockIdx.x * blockDim.x + threadIdx.x;
  if (t >= total) return;
  const int e    = t & 15;
  const int lane = (t >> 4) & 31;
  const int tile = t >> 9;
  const int ktiles = cinp >> 5;
  const int kt = tile % ktiles;
  const int nt = tile / ktiles;
  const int v = e >> 1, h = e & 1;
  const int kbase = (v & 4) ? 16 : 0;
  const int khalf = lane >> 4;
  const int k = kt * 32 + kbase + khalf * 8 + (v & 3) * 2 + h;
  const int n = nt * 16 + (lane & 15);
  Wp[t] = (k < cin) ? (_Float16)W[(size_t)k * cout + n] : (_Float16)0.0f;
}

// ---------------------------------------------------------------------------
// Farthest point sampling: one block per cloud, LDS min-distance array,
// block-wide argmax reduction per iteration.
// ---------------------------------------------------------------------------
__global__ void __launch_bounds__(256) fps_kernel(const float* __restrict__ pos,
                                                  int Pn, int m, int* __restrict__ outIdx) {
  __shared__ float dists[4096];
  __shared__ float rv[256];
  __shared__ int   ri[256];
  __shared__ int   sLast;
  const int b   = blockIdx.x;
  const int tid = threadIdx.x;
  const float* p = pos + (size_t)b * Pn * 3;

  for (int i = tid; i < Pn; i += 256) dists[i] = 1e10f;
  if (tid == 0) { outIdx[(size_t)b * m] = 0; sLast = 0; }
  __syncthreads();

  for (int it = 1; it < m; ++it) {
    const int last = sLast;
    const float lx = p[last * 3 + 0], ly = p[last * 3 + 1], lz = p[last * 3 + 2];
    float bd = -1.0f; int bix = 0;
    for (int i = tid; i < Pn; i += 256) {
      const float dx = p[i * 3 + 0] - lx;
      const float dy = p[i * 3 + 1] - ly;
      const float dz = p[i * 3 + 2] - lz;
      const float nd = fminf(dists[i], dx * dx + dy * dy + dz * dz);
      dists[i] = nd;
      if (nd > bd) { bd = nd; bix = i; }
    }
    rv[tid] = bd; ri[tid] = bix;
    __syncthreads();
    for (int s = 128; s > 0; s >>= 1) {
      if (tid < s && rv[tid + s] > rv[tid]) { rv[tid] = rv[tid + s]; ri[tid] = ri[tid + s]; }
      __syncthreads();
    }
    if (tid == 0) { sLast = ri[0]; outIdx[(size_t)b * m + it] = ri[0]; }
    __syncthreads();
  }
}

__global__ void gather_pos_kernel(const float* __restrict__ pos, const int* __restrict__ idx,
                                  float* __restrict__ out, int m, int Ps) {
  int t = blockIdx.x * blockDim.x + threadIdx.x;
  if (t >= BB * m) return;
  const int b = t / m;
  const int j = idx[t];
  const float* s = pos + ((size_t)b * Ps + j) * 3;
  out[(size_t)t * 3 + 0] = s[0];
  out[(size_t)t * 3 + 1] = s[1];
  out[(size_t)t * 3 + 2] = s[2];
}

// ---------------------------------------------------------------------------
// Brute-force kNN: thread per query, sorted top-K insertion (ties keep the
// earlier source index, matching top_k ordering).
// ---------------------------------------------------------------------------
template <int K>
__global__ void knn_kernel(const float* __restrict__ src, int Ps,
                           const float* __restrict__ dst, int Pd,
                           int* __restrict__ nbr) {
  int t = blockIdx.x * blockDim.x + threadIdx.x;
  if (t >= BB * Pd) return;
  const int b = t / Pd;
  const float* S = src + (size_t)b * Ps * 3;
  const float qx = dst[(size_t)t * 3 + 0];
  const float qy = dst[(size_t)t * 3 + 1];
  const float qz = dst[(size_t)t * 3 + 2];
  float bd[K]; int bi[K];
#pragma unroll
  for (int k = 0; k < K; ++k) { bd[k] = 3e38f; bi[k] = 0; }
  for (int j = 0; j < Ps; ++j) {
    const float dx = S[j * 3 + 0] - qx;
    const float dy = S[j * 3 + 1] - qy;
    const float dz = S[j * 3 + 2] - qz;
    const float d2 = dx * dx + dy * dy + dz * dz;
    if (d2 < bd[K - 1]) {
      int p = K - 1;
      while (p > 0 && bd[p - 1] > d2) { bd[p] = bd[p - 1]; bi[p] = bi[p - 1]; --p; }
      bd[p] = d2; bi[p] = j;
    }
  }
#pragma unroll
  for (int k = 0; k < K; ++k) nbr[(size_t)t * K + k] = bi[k];
}

// ---------------------------------------------------------------------------
// Fused SA module: gather edge msgs -> 3 WMMA MLP layers in LDS -> max-pool.
// 128 threads (4 waves x 16-row stripes), 64 edge rows (2 points) per block.
// ---------------------------------------------------------------------------
template <int CF, int C0P, int C1, int C2, int C3>
__global__ void __launch_bounds__(128) sa_fused_kernel(
    const float* __restrict__ posSrc, const float* __restrict__ featSrc,
    const float* __restrict__ posD, const int* __restrict__ nbr,
    int mC, int PsC,
    const _Float16* __restrict__ W0, const float* __restrict__ b0,
    const _Float16* __restrict__ W1, const float* __restrict__ b1,
    const _Float16* __restrict__ W2, const float* __restrict__ b2,
    float* __restrict__ outFeat) {
  extern __shared__ _Float16 smem[];
  constexpr int AW = (C0P > C2) ? C0P : C2;
  _Float16* bufA = smem;                 // 64 x AW
  _Float16* bufB = smem + 64 * AW;       // 64 x max(C1,C3)

  const int b   = blockIdx.y;
  const int p0  = blockIdx.x * 2;
  const int tid = threadIdx.x;

  // Stage edge messages: [feat[nbr] || pos[nbr]-pos_d] zero-padded to C0P.
  for (int e = tid; e < 64 * C0P; e += 128) {
    const int r = e / C0P, c = e % C0P;
    const int p = p0 + (r >> 5);
    const int j = nbr[((size_t)b * mC + p) * KSA + (r & 31)];
    float v;
    if (c < CF) {
      v = featSrc[((size_t)b * PsC + j) * CF + c];
    } else if (c < CF + 3) {
      v = posSrc[((size_t)b * PsC + j) * 3 + (c - CF)] -
          posD[((size_t)b * mC + p) * 3 + (c - CF)];
    } else {
      v = 0.0f;
    }
    bufA[(size_t)r * C0P + c] = (_Float16)v;
  }
  __syncthreads();

  const int row0 = (tid >> 5) * 16;
  mlp_layer_lds<C0P, C1>(bufA, row0, W0, b0, true, bufB);
  __syncthreads();
  mlp_layer_lds<C1, C2>(bufB, row0, W1, b1, true, bufA);
  __syncthreads();
  mlp_layer_lds<C2, C3>(bufA, row0, W2, b2, false, bufB);
  __syncthreads();

  // Max over the 32 edges of each sampled point.
  for (int o = tid; o < 2 * C3; o += 128) {
    const int pt = o / C3, c = o % C3;
    float mx = -3e38f;
    for (int r = 0; r < 32; ++r)
      mx = fmaxf(mx, (float)bufB[(size_t)(pt * 32 + r) * C3 + c]);
    outFeat[((size_t)b * mC + p0 + pt) * C3 + c] = mx;
  }
}

// ---------------------------------------------------------------------------
// Fused FP module: inverse-d^2 interpolation + concat skip -> 2 WMMA layers.
// 64 threads (2 waves x 16-row stripes), 32 fine points per block; final
// layer streams f32 directly to global.
// ---------------------------------------------------------------------------
template <int CC, int CS, int C0P, int C1, int C2>
__global__ void __launch_bounds__(64) fp_fused_kernel(
    const float* __restrict__ posC, const float* __restrict__ xC, int Pc,
    const float* __restrict__ posF, const float* __restrict__ xS, int Pf,
    const int* __restrict__ nbr,
    const _Float16* __restrict__ W0, const float* __restrict__ b0,
    const _Float16* __restrict__ W1, const float* __restrict__ b1,
    float* __restrict__ outY) {
  extern __shared__ _Float16 smem[];
  _Float16* bufA = smem;                 // 32 x C0P
  _Float16* bufB = smem + 32 * C0P;      // 32 x C1
  __shared__ float wls[32][3];
  __shared__ int   ils[32][3];

  const int b   = blockIdx.y;
  const int r0  = blockIdx.x * 32;
  const int tid = threadIdx.x;

  if (tid < 32) {  // wave-uniform branch: wave0 all-true, wave1 all-false
    const int i = r0 + tid;
    const float px = posF[((size_t)b * Pf + i) * 3 + 0];
    const float py = posF[((size_t)b * Pf + i) * 3 + 1];
    const float pz = posF[((size_t)b * Pf + i) * 3 + 2];
    float ws[3]; float sum = 0.0f;
#pragma unroll
    for (int q = 0; q < 3; ++q) {
      const int j = nbr[((size_t)b * Pf + i) * 3 + q];
      ils[tid][q] = j;
      const float dx = px - posC[((size_t)b * Pc + j) * 3 + 0];
      const float dy = py - posC[((size_t)b * Pc + j) * 3 + 1];
      const float dz = pz - posC[((size_t)b * Pc + j) * 3 + 2];
      const float w = 1.0f / (dx * dx + dy * dy + dz * dz + 1e-8f);
      ws[q] = w; sum += w;
    }
    const float inv = 1.0f / (sum + 1e-8f);
#pragma unroll
    for (int q = 0; q < 3; ++q) wls[tid][q] = ws[q] * inv;
  }
  __syncthreads();

  for (int e = tid; e < 32 * C0P; e += 64) {
    const int r = e / C0P, c = e % C0P;
    float v;
    if (c < CC) {
      v = wls[r][0] * xC[((size_t)b * Pc + ils[r][0]) * CC + c] +
          wls[r][1] * xC[((size_t)b * Pc + ils[r][1]) * CC + c] +
          wls[r][2] * xC[((size_t)b * Pc + ils[r][2]) * CC + c];
    } else if (c < CC + CS) {
      v = xS[((size_t)b * Pf + r0 + r) * CS + (c - CC)];
    } else {
      v = 0.0f;
    }
    bufA[(size_t)r * C0P + c] = (_Float16)v;
  }
  __syncthreads();

  const int row0 = (tid >> 5) * 16;
  mlp_layer_lds<C0P, C1>(bufA, row0, W0, b0, true, bufB);
  __syncthreads();
  mlp_layer_out<C1, C2>(bufB, row0, W1, b1, outY + ((size_t)b * Pf + r0) * C2);
}

// ---------------------------------------------------------------------------
// Host orchestration.
// ---------------------------------------------------------------------------
extern "C" void kernel_launch(void* const* d_in, const int* in_sizes, int n_in,
                              void* d_out, int out_size, void* d_ws, size_t ws_size,
                              hipStream_t stream) {
  (void)in_sizes; (void)n_in; (void)out_size; (void)ws_size;

  const float* pos0 = (const float*)d_in[0];
  // d_in[1] = batch (arange//P -> implied by layout; unused)
  const float* Wf[15]; const float* Bf[15];
  for (int i = 0; i < 15; ++i) { Wf[i] = (const float*)d_in[2 + 2 * i]; Bf[i] = (const float*)d_in[3 + 2 * i]; }

  char* base = (char*)d_ws;
  size_t off = 0;
  auto alloc = [&](size_t bytes) -> void* {
    off = (off + 255) & ~(size_t)255;
    void* p = base + off;
    off += bytes;
    return p;
  };

  _Float16* Wp[15];
  for (int i = 0; i < 15; ++i)
    Wp[i] = (_Float16*)alloc((size_t)L_cinp[i] * L_cout[i] * sizeof(_Float16));
  int*   idx1  = (int*)alloc((size_t)BB * M1 * 4);
  int*   idx2  = (int*)alloc((size_t)BB * M2 * 4);
  int*   idx3  = (int*)alloc((size_t)BB * M3 * 4);
  float* pos1  = (float*)alloc((size_t)BB * M1 * 3 * 4);
  float* pos2  = (float*)alloc((size_t)BB * M2 * 3 * 4);
  float* pos3  = (float*)alloc((size_t)BB * M3 * 3 * 4);
  int*   nbrS1 = (int*)alloc((size_t)BB * M1 * KSA * 4);
  int*   nbrS2 = (int*)alloc((size_t)BB * M2 * KSA * 4);
  int*   nbrS3 = (int*)alloc((size_t)BB * M3 * KSA * 4);
  int*   nbrF3 = (int*)alloc((size_t)BB * M2 * KFP * 4);
  int*   nbrF2 = (int*)alloc((size_t)BB * M1 * KFP * 4);
  int*   nbrF1 = (int*)alloc((size_t)BB * PP * KFP * 4);
  float* x1    = (float*)alloc((size_t)BB * M1 * 128 * 4);
  float* x2    = (float*)alloc((size_t)BB * M2 * 256 * 4);
  float* x3    = (float*)alloc((size_t)BB * M3 * 512 * 4);
  float* y2    = (float*)alloc((size_t)BB * M2 * 256 * 4);
  float* y1    = (float*)alloc((size_t)BB * M1 * 256 * 4);

  // Raise dynamic-LDS caps for the big fused kernels (320KB/WGP budget).
  hipFuncSetAttribute(reinterpret_cast<const void*>(sa_fused_kernel<256, 288, 256, 256, 512>),
                      hipFuncAttributeMaxDynamicSharedMemorySize, 102400);
  hipFuncSetAttribute(reinterpret_cast<const void*>(sa_fused_kernel<128, 160, 128, 128, 256>),
                      hipFuncAttributeMaxDynamicSharedMemorySize, 53248);
  hipFuncSetAttribute(reinterpret_cast<const void*>(fp_fused_kernel<512, 256, 768, 256, 256>),
                      hipFuncAttributeMaxDynamicSharedMemorySize, 65536);

  // Weight conversion (f32 row-major -> f16 WMMA-fragment-major, zero-padded).
  for (int i = 0; i < 15; ++i) {
    const int total = L_cinp[i] * L_cout[i];
    wconv_kernel<<<(total + 255) / 256, 256, 0, stream>>>(Wf[i], Wp[i],
                                                          L_cin[i], L_cout[i], L_cinp[i]);
  }

  // ---- SA1: 4096 -> 1024, MLP [6,64,64,128] ----
  fps_kernel<<<BB, 256, 0, stream>>>(pos0, PP, M1, idx1);
  gather_pos_kernel<<<(BB * M1 + 127) / 128, 128, 0, stream>>>(pos0, idx1, pos1, M1, PP);
  knn_kernel<KSA><<<(BB * M1 + 127) / 128, 128, 0, stream>>>(pos0, PP, pos1, M1, nbrS1);
  sa_fused_kernel<3, 32, 64, 64, 128><<<dim3(M1 / 2, BB), 128, 24576, stream>>>(
      pos0, pos0, pos1, nbrS1, M1, PP, Wp[0], Bf[0], Wp[1], Bf[1], Wp[2], Bf[2], x1);

  // ---- SA2: 1024 -> 256, MLP [131,128,128,256] ----
  fps_kernel<<<BB, 256, 0, stream>>>(pos1, M1, M2, idx2);
  gather_pos_kernel<<<(BB * M2 + 127) / 128, 128, 0, stream>>>(pos1, idx2, pos2, M2, M1);
  knn_kernel<KSA><<<(BB * M2 + 127) / 128, 128, 0, stream>>>(pos1, M1, pos2, M2, nbrS2);
  sa_fused_kernel<128, 160, 128, 128, 256><<<dim3(M2 / 2, BB), 128, 53248, stream>>>(
      pos1, x1, pos2, nbrS2, M2, M1, Wp[3], Bf[3], Wp[4], Bf[4], Wp[5], Bf[5], x2);

  // ---- SA3: 256 -> 64, MLP [259,256,256,512] ----
  fps_kernel<<<BB, 256, 0, stream>>>(pos2, M2, M3, idx3);
  gather_pos_kernel<<<(BB * M3 + 127) / 128, 128, 0, stream>>>(pos2, idx3, pos3, M3, M2);
  knn_kernel<KSA><<<(BB * M3 + 127) / 128, 128, 0, stream>>>(pos2, M2, pos3, M3, nbrS3);
  sa_fused_kernel<256, 288, 256, 256, 512><<<dim3(M3 / 2, BB), 128, 102400, stream>>>(
      pos2, x2, pos3, nbrS3, M3, M2, Wp[6], Bf[6], Wp[7], Bf[7], Wp[8], Bf[8], x3);

  // ---- FP3: interp x3 (64 pts) onto 256 pts, concat x2, MLP [768,256,256] ----
  knn_kernel<KFP><<<(BB * M2 + 127) / 128, 128, 0, stream>>>(pos3, M3, pos2, M2, nbrF3);
  fp_fused_kernel<512, 256, 768, 256, 256><<<dim3(M2 / 32, BB), 64, 65536, stream>>>(
      pos3, x3, M3, pos2, x2, M2, nbrF3, Wp[9], Bf[9], Wp[10], Bf[10], y2);

  // ---- FP2: interp y2 (256 pts) onto 1024 pts, concat x1, MLP [384,256,256] ----
  knn_kernel<KFP><<<(BB * M1 + 127) / 128, 128, 0, stream>>>(pos2, M2, pos1, M1, nbrF2);
  fp_fused_kernel<256, 128, 384, 256, 256><<<dim3(M1 / 32, BB), 64, 40960, stream>>>(
      pos2, y2, M2, pos1, x1, M1, nbrF2, Wp[11], Bf[11], Wp[12], Bf[12], y1);

  // ---- FP1: interp y1 (1024 pts) onto 4096 pts, concat raw pos, MLP [259,256,256] ----
  knn_kernel<KFP><<<(BB * PP + 127) / 128, 128, 0, stream>>>(pos1, M1, pos0, PP, nbrF1);
  fp_fused_kernel<256, 3, 288, 256, 256><<<dim3(PP / 32, BB), 64, 34816, stream>>>(
      pos1, y1, M1, pos0, pos0, PP, nbrF1, Wp[13], Bf[13], Wp[14], Bf[14], (float*)d_out);
}
